// GAE_38946763440858
// MI455X (gfx1250) — matrix operand order repaired and checked by reading, not compiled, not run on hardware
//
#include <hip/hip_runtime.h>
#include <cstddef>

// ---------------------------------------------------------------------------
// Types for CDNA5 WMMA (wave32, 16x16x32 bf16 -> f32 accumulate)
// ---------------------------------------------------------------------------
typedef __bf16 bf16_t;
typedef __attribute__((ext_vector_type(16))) __bf16 v16bf;
typedef __attribute__((ext_vector_type(8)))  __bf16 v8bf;
typedef __attribute__((ext_vector_type(8)))  float  v8f;

#define NN 8192

// ---------------------------------------------------------------------------
// Elementwise cast f32 -> bf16 (vector of 4)
// ---------------------------------------------------------------------------
__global__ void cast_f32_bf16(const float* __restrict__ in,
                              bf16_t* __restrict__ out, size_t n) {
  size_t i = ((size_t)blockIdx.x * blockDim.x + threadIdx.x) * 4;
  if (i + 3 < n) {
    float4 v = *(const float4*)(in + i);
    out[i + 0] = (bf16_t)v.x;
    out[i + 1] = (bf16_t)v.y;
    out[i + 2] = (bf16_t)v.z;
    out[i + 3] = (bf16_t)v.w;
  }
}

// ---------------------------------------------------------------------------
// Tiled transpose + cast: in f32 [R,C] row-major -> out bf16 [C,R] row-major.
// R, C multiples of 32. block = (32,8), grid = (C/32, R/32).
// ---------------------------------------------------------------------------
__global__ void transpose_cast(const float* __restrict__ in,
                               bf16_t* __restrict__ out, int R, int C) {
  __shared__ bf16_t tile[32][33];
  const int c0 = blockIdx.x * 32;
  const int r0 = blockIdx.y * 32;
  for (int dy = threadIdx.y; dy < 32; dy += 8) {
    tile[dy][threadIdx.x] = (bf16_t)in[(size_t)(r0 + dy) * C + (c0 + threadIdx.x)];
  }
  __syncthreads();
  for (int dy = threadIdx.y; dy < 32; dy += 8) {
    out[(size_t)(c0 + dy) * R + (r0 + threadIdx.x)] = tile[threadIdx.x][dy];
  }
}

// ---------------------------------------------------------------------------
// f1[i] = dot(Wh[i,:], a1), f2[i] = dot(Wh[i,:], a2).  One wave32 per row.
// ---------------------------------------------------------------------------
__global__ void attn_vectors(const float* __restrict__ Wh,
                             const float* __restrict__ a1,
                             const float* __restrict__ a2,
                             float* __restrict__ f1, float* __restrict__ f2,
                             int K) {
  const int row  = blockIdx.x * (blockDim.x >> 5) + (threadIdx.x >> 5);
  const int lane = threadIdx.x & 31;
  const float* w = Wh + (size_t)row * K;
  float s1 = 0.f, s2 = 0.f;
  for (int k = lane; k < K; k += 32) {
    float v = w[k];
    s1 += v * a1[k];
    s2 += v * a2[k];
  }
  #pragma unroll
  for (int off = 16; off > 0; off >>= 1) {
    s1 += __shfl_xor(s1, off, 32);
    s2 += __shfl_xor(s2, off, 32);
  }
  if (lane == 0) { f1[row] = s1; f2[row] = s2; }
}

// ---------------------------------------------------------------------------
// Attention numerator: P[i,j] = exp(sigmoid(A[i,j] * (f1[i] + f2[j]))) in bf16,
// invs[i] = 1 / sum_j P[i,j].  One 256-thread block per row i.
// softmax(sigmoid(.)) needs no max-subtraction: sigmoid in (0,1) -> exp in (1,e).
// ---------------------------------------------------------------------------
__global__ void attn_numerator(const float* __restrict__ A,
                               const float* __restrict__ f1,
                               const float* __restrict__ f2,
                               bf16_t* __restrict__ P,
                               float* __restrict__ invs, int n) {
  const int i = blockIdx.x;
  const float fi = f1[i];
  const float* arow = A + (size_t)i * n;
  bf16_t* prow = P + (size_t)i * n;
  float part = 0.f;
  for (int j0 = threadIdx.x * 8; j0 < n; j0 += blockDim.x * 8) {
    float4 a0 = *(const float4*)(arow + j0);
    float4 a1 = *(const float4*)(arow + j0 + 4);
    float av[8] = {a0.x, a0.y, a0.z, a0.w, a1.x, a1.y, a1.z, a1.w};
    v8bf pk;
    #pragma unroll
    for (int t = 0; t < 8; ++t) {
      float z = av[t] * (fi + f2[j0 + t]);
      float s = 1.0f / (1.0f + __expf(-z));   // sigmoid
      float p = __expf(s);                    // softmax numerator
      part += p;
      pk[t] = (bf16_t)p;
    }
    *(v8bf*)(prow + j0) = pk;
  }
  __shared__ float red[256];
  red[threadIdx.x] = part;
  __syncthreads();
  for (int s = 128; s > 0; s >>= 1) {
    if (threadIdx.x < (unsigned)s) red[threadIdx.x] += red[threadIdx.x + s];
    __syncthreads();
  }
  if (threadIdx.x == 0) invs[i] = 1.0f / red[0];
}

// ---------------------------------------------------------------------------
// bf16 WMMA GEMM:  C[M,N] (f32) = rowscale ⊙ (A[M,K] @ B[K,N]) + bias
//  - A   : bf16 row-major [M,K]
//  - BT  : bf16 row-major [N,K]   (i.e. B stored column-major)
//  - bias: f32 [N] or nullptr;  rowscale: f32 [M] or nullptr
//
// 256 threads = 8 waves. Wave computes a 32x64 C-tile: 2 A fragments x
// 4 B fragments -> 8 WMMAs per 32-wide k-step (1.5 b128 loads per WMMA,
// ~21 FLOP per fragment byte). Block covers 128x128 (4 M-waves x 2 N-waves).
// M % 128 == 0, N % 128 == 0, K % 32 == 0. EXEC is all-ones (no divergence).
//
// With contiguous k order (k = 16h + s), per-lane fragment loads are contiguous:
//  A frag: row (lane&15), two 16B chunks at k+{0,8} (+16 elems for lanes 16-31)
//  B frag: col (lane&15), one 32B chunk at k (+16 elems for lanes 16-31) in BT
// ---------------------------------------------------------------------------
__global__ void gemm_bf16_wmma(const bf16_t* __restrict__ A,
                               const bf16_t* __restrict__ BT,
                               float* __restrict__ C,
                               const float* __restrict__ bias,
                               const float* __restrict__ rowscale,
                               int M, int N, int K) {
  const int wave = threadIdx.x >> 5;
  const int lane = threadIdx.x & 31;
  const int wm = wave >> 1;               // 0..3 : 32-row sub-tile
  const int wn = wave & 1;                // 0..1 : 64-col sub-tile
  const int m0 = blockIdx.y * 128 + wm * 32;
  const int n0 = blockIdx.x * 128 + wn * 64;

  const int am  = lane & 15;
  const int akb = (lane < 16) ? 0 : 16;   // A: K sub-offset (contiguous k order)
  const int bn  = lane & 15;
  const int bko = (lane < 16) ? 0 : 16;   // B: K sub-offset (contiguous k order)

  const bf16_t* arow0 = A + (size_t)(m0 +  0 + am) * K + akb;
  const bf16_t* arow1 = A + (size_t)(m0 + 16 + am) * K + akb;
  const bf16_t* b0r   = BT + (size_t)(n0 +  0 + bn) * K + bko;
  const bf16_t* b1r   = BT + (size_t)(n0 + 16 + bn) * K + bko;
  const bf16_t* b2r   = BT + (size_t)(n0 + 32 + bn) * K + bko;
  const bf16_t* b3r   = BT + (size_t)(n0 + 48 + bn) * K + bko;

  v8f acc00 = {}, acc01 = {}, acc02 = {}, acc03 = {};
  v8f acc10 = {}, acc11 = {}, acc12 = {}, acc13 = {};

  for (int k = 0; k < K; k += 32) {
    v8bf a0lo = *(const v8bf*)(arow0 + k);
    v8bf a0hi = *(const v8bf*)(arow0 + k + 8);
    v8bf a1lo = *(const v8bf*)(arow1 + k);
    v8bf a1hi = *(const v8bf*)(arow1 + k + 8);
    v16bf af0 = __builtin_shufflevector(a0lo, a0hi,
        0, 1, 2, 3, 4, 5, 6, 7, 8, 9, 10, 11, 12, 13, 14, 15);
    v16bf af1 = __builtin_shufflevector(a1lo, a1hi,
        0, 1, 2, 3, 4, 5, 6, 7, 8, 9, 10, 11, 12, 13, 14, 15);
    v16bf bf0 = *(const v16bf*)(b0r + k);
    v16bf bf1 = *(const v16bf*)(b1r + k);
    v16bf bf2 = *(const v16bf*)(b2r + k);
    v16bf bf3 = *(const v16bf*)(b3r + k);
    acc00 = __builtin_amdgcn_wmma_f32_16x16x32_bf16(false, af0, false, bf0,
                                                    (short)0, acc00, false, false);
    acc10 = __builtin_amdgcn_wmma_f32_16x16x32_bf16(false, af1, false, bf0,
                                                    (short)0, acc10, false, false);
    acc01 = __builtin_amdgcn_wmma_f32_16x16x32_bf16(false, af0, false, bf1,
                                                    (short)0, acc01, false, false);
    acc11 = __builtin_amdgcn_wmma_f32_16x16x32_bf16(false, af1, false, bf1,
                                                    (short)0, acc11, false, false);
    acc02 = __builtin_amdgcn_wmma_f32_16x16x32_bf16(false, af0, false, bf2,
                                                    (short)0, acc02, false, false);
    acc12 = __builtin_amdgcn_wmma_f32_16x16x32_bf16(false, af1, false, bf2,
                                                    (short)0, acc12, false, false);
    acc03 = __builtin_amdgcn_wmma_f32_16x16x32_bf16(false, af0, false, bf3,
                                                    (short)0, acc03, false, false);
    acc13 = __builtin_amdgcn_wmma_f32_16x16x32_bf16(false, af1, false, bf3,
                                                    (short)0, acc13, false, false);
  }

  // C/D layout: VGPR r -> row base + r (+8 for lanes 16-31), col = lane&15
  const int rsub  = (lane < 16) ? 0 : 8;
  const int cbase = lane & 15;
  v8f accs[2][4] = {{acc00, acc01, acc02, acc03}, {acc10, acc11, acc12, acc13}};
  #pragma unroll
  for (int h = 0; h < 2; ++h) {
    const int rbase = m0 + h * 16 + rsub;
    #pragma unroll
    for (int t = 0; t < 4; ++t) {
      const int c = n0 + t * 16 + cbase;
      const float bv = bias ? bias[c] : 0.0f;
      #pragma unroll
      for (int i = 0; i < 8; ++i) {
        const int r = rbase + i;
        const float s = rowscale ? rowscale[r] : 1.0f;
        C[(size_t)r * N + c] = accs[h][t][i] * s + bv;
      }
    }
  }
}

// ---------------------------------------------------------------------------
// Host orchestration
// ---------------------------------------------------------------------------
extern "C" void kernel_launch(void* const* d_in, const int* in_sizes, int n_in,
                              void* d_out, int out_size, void* d_ws, size_t ws_size,
                              hipStream_t stream) {
  (void)in_sizes; (void)n_in; (void)out_size; (void)ws_size;
  const int N = NN;

  const float* x     = (const float*)d_in[0];
  const float* A     = (const float*)d_in[1];
  const float* en_w0 = (const float*)d_in[2];
  const float* en_b0 = (const float*)d_in[3];
  const float* en_w1 = (const float*)d_in[4];
  const float* en_b1 = (const float*)d_in[5];
  const float* a1_0  = (const float*)d_in[6];
  const float* a2_0  = (const float*)d_in[7];
  const float* a1_1  = (const float*)d_in[8];
  const float* a2_1  = (const float*)d_in[9];
  const float* de_w0 = (const float*)d_in[10];
  const float* de_b0 = (const float*)d_in[11];
  const float* de_w1 = (const float*)d_in[12];
  const float* de_b1 = (const float*)d_in[13];

  float* out_hidden = (float*)d_out;                       // [8192,128]
  float* out_res    = out_hidden + (size_t)N * 128;        // [8192,512]
  float* out_G      = out_res + (size_t)N * 512;           // [8192,8192]

  // Workspace bump allocator (256B aligned)
  char* wp = (char*)d_ws;
  auto alloc = [&](size_t bytes) -> void* {
    void* r = (void*)wp;
    wp += (bytes + 255) & ~(size_t)255;
    return r;
  };
  bf16_t* x_bf   = (bf16_t*)alloc((size_t)N * 512 * 2);
  bf16_t* w0T    = (bf16_t*)alloc((size_t)512 * 256 * 2);  // [256,512]
  float*  Wh0f   = (float*) alloc((size_t)N * 256 * 4);
  bf16_t* Wh0T   = (bf16_t*)alloc((size_t)N * 256 * 2);    // [256,8192]
  float*  f1_0   = (float*) alloc((size_t)N * 4);
  float*  f2_0   = (float*) alloc((size_t)N * 4);
  bf16_t* P0     = (bf16_t*)alloc((size_t)N * N * 2);
  float*  invs0  = (float*) alloc((size_t)N * 4);
  float*  res0f  = (float*) alloc((size_t)N * 256 * 4);
  bf16_t* res0bf = (bf16_t*)alloc((size_t)N * 256 * 2);
  bf16_t* w1T    = (bf16_t*)alloc((size_t)256 * 128 * 2);  // [128,256]
  float*  Wh1f   = (float*) alloc((size_t)N * 128 * 4);
  bf16_t* Wh1T   = (bf16_t*)alloc((size_t)N * 128 * 2);    // [128,8192]
  float*  f1_1   = (float*) alloc((size_t)N * 4);
  float*  f2_1   = (float*) alloc((size_t)N * 4);
  bf16_t* P1     = (bf16_t*)alloc((size_t)N * N * 2);
  float*  invs1  = (float*) alloc((size_t)N * 4);
  bf16_t* hid_bf = (bf16_t*)alloc((size_t)N * 128 * 2);
  bf16_t* dw0T   = (bf16_t*)alloc((size_t)128 * 256 * 2);  // [256,128]
  float*  D0f    = (float*) alloc((size_t)N * 256 * 4);
  bf16_t* D0T    = (bf16_t*)alloc((size_t)N * 256 * 2);    // [256,8192]
  float*  T1f    = (float*) alloc((size_t)N * 256 * 4);
  bf16_t* T1bf   = (bf16_t*)alloc((size_t)N * 256 * 2);
  bf16_t* dw1T   = (bf16_t*)alloc((size_t)256 * 512 * 2);  // [512,256]
  float*  D1f    = (float*) alloc((size_t)N * 512 * 4);
  bf16_t* D1T    = (bf16_t*)alloc((size_t)N * 512 * 2);    // [512,8192]

  auto cast = [&](const float* in, bf16_t* out, size_t n) {
    cast_f32_bf16<<<dim3((unsigned)((n / 4 + 255) / 256)), dim3(256), 0, stream>>>(in, out, n);
  };
  auto tr = [&](const float* in, bf16_t* out, int R, int Cc) {
    transpose_cast<<<dim3(Cc / 32, R / 32), dim3(32, 8), 0, stream>>>(in, out, R, Cc);
  };
  auto gemm = [&](const bf16_t* Abf, const bf16_t* BT, float* C,
                  const float* bias, const float* rowscale, int M, int Nc, int K) {
    gemm_bf16_wmma<<<dim3(Nc / 128, M / 128), dim3(256), 0, stream>>>(
        Abf, BT, C, bias, rowscale, M, Nc, K);
  };
  auto fvec = [&](const float* Wh, const float* a1, const float* a2,
                  float* f1, float* f2, int K) {
    attn_vectors<<<dim3(N / 8), dim3(256), 0, stream>>>(Wh, a1, a2, f1, f2, K);
  };
  auto attn = [&](const float* f1, const float* f2, bf16_t* P, float* invs) {
    attn_numerator<<<dim3(N), dim3(256), 0, stream>>>(A, f1, f2, P, invs, N);
  };

  // ---- Encoder layer 0 ----
  cast(x, x_bf, (size_t)N * 512);
  tr(en_w0, w0T, 512, 256);
  gemm(x_bf, w0T, Wh0f, en_b0, nullptr, N, 256, 512);       // Wh0 = x@W0 + b0
  fvec(Wh0f, a1_0, a2_0, f1_0, f2_0, 256);
  attn(f1_0, f2_0, P0, invs0);                              // attn0 numerator
  tr(Wh0f, Wh0T, N, 256);
  gemm(P0, Wh0T, res0f, nullptr, invs0, N, 256, N);         // res0 = attn0@Wh0

  // ---- Encoder layer 1 ----
  cast(res0f, res0bf, (size_t)N * 256);
  tr(en_w1, w1T, 256, 128);
  gemm(res0bf, w1T, Wh1f, en_b1, nullptr, N, 128, 256);     // Wh1 = res0@W1 + b1
  fvec(Wh1f, a1_1, a2_1, f1_1, f2_1, 128);
  attn(f1_1, f2_1, P1, invs1);                              // attn1 numerator
  tr(Wh1f, Wh1T, N, 128);
  gemm(P1, Wh1T, out_hidden, nullptr, invs1, N, 128, N);    // hidden = attn1@Wh1

  // ---- Decoder ----
  cast(out_hidden, hid_bf, (size_t)N * 128);
  tr(de_w0, dw0T, 128, 256);
  gemm(hid_bf, dw0T, D0f, de_b0, nullptr, N, 256, 128);     // D0 = hidden@dW0 + b
  tr(D0f, D0T, N, 256);
  gemm(P1, D0T, T1f, nullptr, invs1, N, 256, N);            // T1 = attn1@D0
  cast(T1f, T1bf, (size_t)N * 256);
  tr(de_w1, dw1T, 256, 512);
  gemm(T1bf, dw1T, D1f, de_b1, nullptr, N, 512, 256);       // D1 = T1@dW1 + b
  tr(D1f, D1T, N, 512);
  gemm(P0, D1T, out_res, nullptr, invs0, N, 512, N);        // res = attn0@D1

  // ---- Gram matrix: G = hidden @ hidden^T  (BT of hidden^T is hidden itself)
  gemm(hid_bf, hid_bf, out_G, nullptr, nullptr, N, N, 128);
}
